// CondInst_22565758173944
// MI455X (gfx1250) — matrix operand order repaired and checked by reading, not compiled, not run on hardware
//
#include <hip/hip_runtime.h>
#include <hip/hip_bf16.h>

#define NUM_CLS 80
#define N_INST  100
#define PARAM_DIM 169
#define PBLK 32

typedef __attribute__((ext_vector_type(16))) __bf16 bf16x16;
typedef __attribute__((ext_vector_type(8)))  float  f32x8;
typedef __attribute__((ext_vector_type(4)))  unsigned int u32x4;

union Frag { u32x4 q[2]; bf16x16 v; };

static __device__ __forceinline__ unsigned short f2bf(float f) {
    unsigned u = __float_as_uint(f);
    // round-to-nearest-even fp32 -> bf16
    unsigned r = (u + 0x7FFFu + ((u >> 16) & 1u)) >> 16;
    return (unsigned short)r;
}

// ---------------------------------------------------------------------------
// Implicit-GEMM conv (KSxKS, stride 1, SAME pad) via v_wmma_f32_16x16x32_bf16.
// Inputs: bf16 activations (NCHW) + bf16 weights pre-transposed to [t][co][ci].
// Block tile: 128 Cout x 64 pixels (8x8 patch), 8 waves (4m x 2n),
// each wave 32m x 32n = 4 WMMA accumulators. Activation halo staged once per
// 32-channel slice and reused across all KS*KS taps. Output f32 (+opt bias).
// grid = (ceil(W/8)*ceil(H/8), ceil(Cout/128), N)
// ---------------------------------------------------------------------------
template <int KS>
__global__ __launch_bounds__(256, 6)
void conv_wmma(const unsigned short* __restrict__ in, const unsigned short* __restrict__ w,
               const float* __restrict__ bias, float* __restrict__ out,
               int Cin, int Cout, int H, int W, int nTx)
{
    constexpr int TAPS = KS * KS;
    constexpr int PAD  = KS >> 1;
    constexpr int HS   = 8 + KS - 1;          // halo side
    constexpr int HCELL = HS * HS;
    const int HW  = H * W;
    const int img = blockIdx.z;
    const int co0 = blockIdx.y * 128;
    const int tx  = blockIdx.x % nTx;
    const int ty  = blockIdx.x / nTx;
    const int x0  = tx * 8, y0 = ty * 8;
    const int tid = threadIdx.x;
    const int lane = tid & 31;
    const int wid  = tid >> 5;
    const int wm   = wid >> 1;   // 0..3: 32-wide cout sub-tile
    const int wn   = wid & 1;    // 0..1: 32-wide pixel sub-tile

    __shared__ __align__(16) unsigned short As[128][40];      // [m][k]
    __shared__ __align__(16) unsigned short Hl[HS][HS][40];   // halo [hy][hx][k]

    f32x8 acc00 = {}, acc01 = {}, acc10 = {}, acc11 = {};

    const unsigned short* inN = in + (size_t)img * Cin * HW;

    #pragma unroll 1
    for (int kb = 0; kb < Cin; kb += 32) {
        __syncthreads();
        // ---- stage activation halo: HS x HS pixels x 32 channels (pure copy) ----
        #pragma unroll 1
        for (int e = tid; e < HCELL * 32; e += 256) {
            int cell = e % HCELL;            // consecutive tid -> consecutive x
            int k    = e / HCELL;
            int hy = cell / HS, hx = cell % HS;
            int sy = y0 + hy - PAD, sx = x0 + hx - PAD;
            unsigned short v = 0;
            if ((unsigned)sy < (unsigned)H && (unsigned)sx < (unsigned)W)
                v = inN[(size_t)(kb + k) * HW + (size_t)sy * W + sx];
            Hl[hy][hx][k] = v;
        }
        if (kb + 32 < Cin)
            __builtin_prefetch(&inN[(size_t)(kb + 32) * HW + (size_t)y0 * W + x0], 0, 1);

        #pragma unroll 1
        for (int t = 0; t < TAPS; ++t) {
            __syncthreads();   // halo ready / previous tap's A reads done
            // ---- stage A (weights): 128 cout x 32 cin, vectorized b128 copies ----
            #pragma unroll
            for (int j = 0; j < 2; ++j) {
                int e8 = j * 256 + tid;      // 512 octets of 8 ushorts
                int m  = e8 >> 2;
                int ko = (e8 & 3) * 8;
                int co = co0 + m;
                u32x4 val = {};
                if (co < Cout)
                    val = *(const u32x4*)&w[((size_t)t * Cout + co) * Cin + kb + ko];
                *(u32x4*)&As[m][ko] = val;
            }
            __syncthreads();

            // ---- fragments ----
            Frag fa0, fa1, fb0, fb1;
            const int mr = wm * 32 + (lane & 15);
            const int kh = lane >> 4;                 // half-wave selector
            fa0.q[0] = *(const u32x4*)&As[mr][kh * 8];       // A: K {0..7 | 8..15}
            fa0.q[1] = *(const u32x4*)&As[mr][16 + kh * 8];  //    K {16..23 | 24..31}
            fa1.q[0] = *(const u32x4*)&As[mr + 16][kh * 8];
            fa1.q[1] = *(const u32x4*)&As[mr + 16][16 + kh * 8];

            const int tyk = t / KS, txk = t % KS;     // 0-based tap offsets
            const int n0 = wn * 32 + (lane & 15);
            const int n1 = n0 + 16;
            const int kb16 = kh * 8;                  // B: K {0..7 | 8..15}, {16..23 | 24..31} per half-wave
            const int py0p = (n0 >> 3) + tyk, px0p = (n0 & 7) + txk;
            const int py1p = (n1 >> 3) + tyk, px1p = (n1 & 7) + txk;
            fb0.q[0] = *(const u32x4*)&Hl[py0p][px0p][kb16];
            fb0.q[1] = *(const u32x4*)&Hl[py0p][px0p][kb16 + 16];
            fb1.q[0] = *(const u32x4*)&Hl[py1p][px1p][kb16];
            fb1.q[1] = *(const u32x4*)&Hl[py1p][px1p][kb16 + 16];

            acc00 = __builtin_amdgcn_wmma_f32_16x16x32_bf16(false, fa0.v, false, fb0.v,
                                                            (short)0, acc00, false, false);
            acc01 = __builtin_amdgcn_wmma_f32_16x16x32_bf16(false, fa0.v, false, fb1.v,
                                                            (short)0, acc01, false, false);
            acc10 = __builtin_amdgcn_wmma_f32_16x16x32_bf16(false, fa1.v, false, fb0.v,
                                                            (short)0, acc10, false, false);
            acc11 = __builtin_amdgcn_wmma_f32_16x16x32_bf16(false, fa1.v, false, fb1.v,
                                                            (short)0, acc11, false, false);
        }
    }

    // ---- epilogue: C/D layout (vgpr r, lane<16 -> M=r, lane>=16 -> M=8+r; N=lane&15)
    const int nlo = lane & 15;
    const int mhi = (lane >> 4) * 8;
    #pragma unroll
    for (int mi = 0; mi < 2; ++mi) {
        #pragma unroll
        for (int r = 0; r < 8; ++r) {
            int co = co0 + wm * 32 + mi * 16 + mhi + r;
            if (co < Cout) {
                float bv = bias ? bias[co] : 0.f;
                size_t ob = ((size_t)img * Cout + co) * HW;
                #pragma unroll
                for (int ni = 0; ni < 2; ++ni) {
                    int n  = wn * 32 + ni * 16 + nlo;
                    int yy = y0 + (n >> 3);
                    int xx = x0 + (n & 7);
                    if (yy < H && xx < W) {
                        float val;
                        if (mi == 0) val = (ni == 0) ? acc00[r] : acc01[r];
                        else         val = (ni == 0) ? acc10[r] : acc11[r];
                        out[ob + (size_t)yy * W + xx] = val + bv;
                    }
                }
            }
        }
    }
}

// ---------------------------------------------------------------------------
// One-time converts: f32 -> bf16 elementwise, and OIHW f32 weights ->
// [tap][co][ci] bf16.
// ---------------------------------------------------------------------------
__global__ __launch_bounds__(256)
void convert_bf16(const float* __restrict__ x, unsigned short* __restrict__ o, size_t total)
{
    size_t stride = (size_t)gridDim.x * blockDim.x;
    for (size_t i = (size_t)blockIdx.x * blockDim.x + threadIdx.x; i < total; i += stride)
        o[i] = f2bf(x[i]);
}

__global__ __launch_bounds__(256)
void convert_w(const float* __restrict__ w, unsigned short* __restrict__ o,
               int Cout, int Cin, int taps)
{
    size_t total = (size_t)taps * Cout * Cin;
    size_t stride = (size_t)gridDim.x * blockDim.x;
    for (size_t i = (size_t)blockIdx.x * blockDim.x + threadIdx.x; i < total; i += stride) {
        int t   = (int)(i / ((size_t)Cout * Cin));
        int rem = (int)(i % ((size_t)Cout * Cin));
        int co = rem / Cin, ci = rem % Cin;
        o[i] = f2bf(w[((size_t)co * Cin + ci) * taps + t]);
    }
}

// ---------------------------------------------------------------------------
// GroupNorm: deterministic three-pass (partials -> finalize -> apply+ReLU)
// ---------------------------------------------------------------------------
__global__ __launch_bounds__(256)
void gn_partial(const float* __restrict__ x, float* __restrict__ part,
                int C, int HW, int groups, int cpg)
{
    int gidx = blockIdx.x / PBLK;
    int pb   = blockIdx.x % PBLK;
    int n = gidx / groups, grp = gidx % groups;
    size_t base = ((size_t)n * C + (size_t)grp * cpg) * HW;
    size_t len  = (size_t)cpg * HW;
    float s = 0.f, ss = 0.f;
    for (size_t i = (size_t)pb * blockDim.x + threadIdx.x; i < len;
         i += (size_t)PBLK * blockDim.x) {
        float v = x[base + i];
        s += v; ss += v * v;
    }
    __shared__ float sh1[256], sh2[256];
    sh1[threadIdx.x] = s; sh2[threadIdx.x] = ss;
    __syncthreads();
    for (int off = 128; off; off >>= 1) {
        if ((int)threadIdx.x < off) {
            sh1[threadIdx.x] += sh1[threadIdx.x + off];
            sh2[threadIdx.x] += sh2[threadIdx.x + off];
        }
        __syncthreads();
    }
    if (threadIdx.x == 0) {
        part[(size_t)blockIdx.x * 2]     = sh1[0];
        part[(size_t)blockIdx.x * 2 + 1] = sh2[0];
    }
}

__global__ void gn_finalize(const float* __restrict__ part, float* __restrict__ mr, float len)
{
    __shared__ float s1[32], s2[32];
    int t = threadIdx.x;
    s1[t] = part[(size_t)(blockIdx.x * PBLK + t) * 2];
    s2[t] = part[(size_t)(blockIdx.x * PBLK + t) * 2 + 1];
    __syncthreads();
    for (int off = 16; off; off >>= 1) {
        if (t < off) { s1[t] += s1[t + off]; s2[t] += s2[t + off]; }
        __syncthreads();
    }
    if (t == 0) {
        float mean = s1[0] / len;
        float var  = s2[0] / len - mean * mean;
        mr[blockIdx.x * 2]     = mean;
        mr[blockIdx.x * 2 + 1] = rsqrtf(var + 1e-5f);
    }
}

__global__ __launch_bounds__(256)
void gn_apply_f32(const float* __restrict__ x, float* __restrict__ out,
                  const float* __restrict__ gamma, const float* __restrict__ beta,
                  const float* __restrict__ mr, int C, int HW, int groups, int cpg,
                  size_t total)
{
    size_t stride = (size_t)gridDim.x * blockDim.x;
    for (size_t i = (size_t)blockIdx.x * blockDim.x + threadIdx.x; i < total; i += stride) {
        size_t c = (i / HW) % C;
        size_t n = i / ((size_t)C * HW);
        int g = (int)(n * groups + c / cpg);
        float v = (x[i] - mr[g * 2]) * mr[g * 2 + 1] * gamma[c] + beta[c];
        out[i] = fmaxf(v, 0.f);
    }
}

// bf16-emitting variant; optional post-scale (applied after ReLU, per reference)
__global__ __launch_bounds__(256)
void gn_apply_bf16(const float* __restrict__ x, unsigned short* __restrict__ out,
                   const float* __restrict__ gamma, const float* __restrict__ beta,
                   const float* __restrict__ mr, const float* __restrict__ post_scale,
                   int C, int HW, int groups, int cpg, size_t total)
{
    float ps = post_scale ? post_scale[0] : 1.0f;
    size_t stride = (size_t)gridDim.x * blockDim.x;
    for (size_t i = (size_t)blockIdx.x * blockDim.x + threadIdx.x; i < total; i += stride) {
        size_t c = (i / HW) % C;
        size_t n = i / ((size_t)C * HW);
        int g = (int)(n * groups + c / cpg);
        float v = (x[i] - mr[g * 2]) * mr[g * 2 + 1] * gamma[c] + beta[c];
        v = fmaxf(v, 0.f) * ps;
        out[i] = f2bf(v);
    }
}

// ---------------------------------------------------------------------------
// Align-corners bilinear resize; addmode=1 accumulates into dst.
// ---------------------------------------------------------------------------
__global__ __launch_bounds__(256)
void resize_bilinear(const float* __restrict__ src, float* __restrict__ dst,
                     int NC, int Hs, int Ws, int Hd, int Wd, int addmode)
{
    size_t total = (size_t)NC * Hd * Wd;
    float ry = (Hd > 1) ? (float)(Hs - 1) / (float)(Hd - 1) : 0.f;
    float rx = (Wd > 1) ? (float)(Ws - 1) / (float)(Wd - 1) : 0.f;
    size_t stride = (size_t)gridDim.x * blockDim.x;
    for (size_t i = (size_t)blockIdx.x * blockDim.x + threadIdx.x; i < total; i += stride) {
        int ox = (int)(i % Wd);
        int oy = (int)((i / Wd) % Hd);
        size_t nc = i / ((size_t)Wd * Hd);
        float fy = oy * ry; int y0 = (int)floorf(fy); int y1 = min(y0 + 1, Hs - 1);
        float wy = fy - (float)y0;
        float fx = ox * rx; int x0 = (int)floorf(fx); int x1 = min(x0 + 1, Ws - 1);
        float wx = fx - (float)x0;
        const float* s = src + nc * (size_t)Hs * Ws;
        float tl = s[(size_t)y0 * Ws + x0], bl = s[(size_t)y1 * Ws + x0];
        float tr = s[(size_t)y0 * Ws + x1], br = s[(size_t)y1 * Ws + x1];
        float v = (tl * (1.f - wy) + bl * wy) * (1.f - wx) +
                  (tr * (1.f - wy) + br * wy) * wx;
        if (addmode) dst[i] += v; else dst[i] = v;
    }
}

// ---------------------------------------------------------------------------
// Per-instance dynamic conv: 10 -> 8 -> 8 -> 1 MLP over (mask_feats, coords).
// Params come straight out of preds[0] (channels 85..253, first 100 pixels).
// grid.x = B * N_INST
// ---------------------------------------------------------------------------
__global__ __launch_bounds__(256)
void dyn_conv(const float* __restrict__ mask_feats, const float* __restrict__ pred0,
              float* __restrict__ out, int H, int W)
{
    const int HW = H * W;
    const int inst = blockIdx.x;
    const int b = inst / N_INST, j = inst % N_INST;
    __shared__ float p[PARAM_DIM];
    for (int i = threadIdx.x; i < PARAM_DIM; i += blockDim.x)
        p[i] = pred0[((size_t)b * 254 + (NUM_CLS + 5) + i) * HW + j];
    __syncthreads();

    for (int px = threadIdx.x; px < HW; px += blockDim.x) {
        int y = px / W, x = px - y * W;
        float in10[10];
        #pragma unroll
        for (int c = 0; c < 8; ++c)
            in10[c] = mask_feats[((size_t)b * 8 + c) * HW + px];
        in10[8] = -1.f + 2.f * (float)x / (float)(W - 1);   // xx
        in10[9] = -1.f + 2.f * (float)y / (float)(H - 1);   // yy
        float h1[8];
        #pragma unroll
        for (int o = 0; o < 8; ++o) {
            float a = p[152 + o];
            #pragma unroll
            for (int c = 0; c < 10; ++c) a += p[o * 10 + c] * in10[c];
            h1[o] = fmaxf(a, 0.f);
        }
        float h2[8];
        #pragma unroll
        for (int o = 0; o < 8; ++o) {
            float a = p[160 + o];
            #pragma unroll
            for (int c = 0; c < 8; ++c) a += p[80 + o * 8 + c] * h1[c];
            h2[o] = fmaxf(a, 0.f);
        }
        float r = p[168];
        #pragma unroll
        for (int c = 0; c < 8; ++c) r += p[144 + c] * h2[c];
        out[(size_t)inst * HW + px] = r;
    }
}

// ---------------------------------------------------------------------------
extern "C" void kernel_launch(void* const* d_in, const int* in_sizes, int n_in,
                              void* d_out, int out_size, void* d_ws, size_t ws_size,
                              hipStream_t stream)
{
    (void)in_sizes; (void)n_in; (void)out_size; (void)ws_size;

    const float* feat[5] = {(const float*)d_in[0], (const float*)d_in[1],
                            (const float*)d_in[2], (const float*)d_in[3],
                            (const float*)d_in[4]};
    const int Hh[5] = {96, 48, 24, 12, 6};
    const int Wz[5] = {160, 80, 40, 20, 10};
    const int hw[5] = {15360, 3840, 960, 240, 60};
    const float* head_w     = (const float*)d_in[5];
    const float* head_gn_g  = (const float*)d_in[6];
    const float* head_gn_b  = (const float*)d_in[7];
    const float* head_out_w = (const float*)d_in[8];
    const float* head_out_b = (const float*)d_in[9];
    const float* scales     = (const float*)d_in[10];
    const float* ref_w      = (const float*)d_in[11];
    const float* ref_gn_g   = (const float*)d_in[12];
    const float* ref_gn_b   = (const float*)d_in[13];
    const float* tow_w      = (const float*)d_in[14];
    const float* tow_gn_g   = (const float*)d_in[15];
    const float* tow_gn_b   = (const float*)d_in[16];
    const float* out_w      = (const float*)d_in[17];
    const float* out_gn_g   = (const float*)d_in[18];
    const float* out_gn_b   = (const float*)d_in[19];

    // ---- workspace layout (bytes, 256B-aligned chunks) ----
    char* wsb = (char*)d_ws;
    size_t off = 0;
    auto alloc = [&](size_t bytes) { size_t c = off; off += (bytes + 255) & ~(size_t)255; return c; };
    const size_t oConvF = alloc((size_t)2 * 256 * 15360 * 4); // f32 conv-out temp (also mask temp)
    const size_t oActBF = alloc((size_t)2 * 256 * 15360 * 2); // bf16 activation temp
    const size_t oFeatBF= alloc((size_t)2 * 256 * 20460 * 2); // bf16 FPN feats (all levels)
    const size_t oMX    = alloc((size_t)2 * 128 * 15360 * 4); // f32 mask accumulation
    const size_t oMFe   = alloc((size_t)2 * 8 * 15360 * 4);   // f32 pre-GN mask feats
    const size_t oML    = alloc((size_t)200 * 15360 * 4);     // f32 pre-upsample logits
    const size_t oWHead = alloc((size_t)4 * 9 * 256 * 256 * 2);
    const size_t oWHOut = alloc((size_t)9 * 254 * 256 * 2);
    const size_t oWRef  = alloc((size_t)3 * 9 * 128 * 256 * 2);
    const size_t oWTow  = alloc((size_t)4 * 9 * 128 * 128 * 2);
    const size_t oWOut  = alloc((size_t)8 * 128 * 2);
    const size_t oPart  = alloc((size_t)64 * PBLK * 2 * 4);
    const size_t oMR    = alloc((size_t)64 * 2 * 4);

    float*          convF  = (float*)(wsb + oConvF);
    unsigned short* actBF  = (unsigned short*)(wsb + oActBF);
    unsigned short* featBF = (unsigned short*)(wsb + oFeatBF);
    float*          mX     = (float*)(wsb + oMX);
    float*          maskT  = convF;                      // reuse (heads finished)
    unsigned short* bfA    = actBF;                      // reuse for tower ping
    unsigned short* bfB    = actBF + (size_t)2 * 128 * 15360; // tower pong
    float*          mFe    = (float*)(wsb + oMFe);
    float*          mLog   = (float*)(wsb + oML);
    unsigned short* wHead  = (unsigned short*)(wsb + oWHead);
    unsigned short* wHOut  = (unsigned short*)(wsb + oWHOut);
    unsigned short* wRef   = (unsigned short*)(wsb + oWRef);
    unsigned short* wTow   = (unsigned short*)(wsb + oWTow);
    unsigned short* wOut   = (unsigned short*)(wsb + oWOut);
    float*          part   = (float*)(wsb + oPart);
    float*          mr     = (float*)(wsb + oMR);

    auto gblocks = [](size_t total) {
        size_t b = (total + 255) / 256;
        return (int)(b < 4096 ? b : 4096);
    };

    // ---- one-time converts: weights (OIHW -> [t][co][ci] bf16), feats -> bf16 ----
    for (int l = 0; l < 4; ++l)
        convert_w<<<2048, 256, 0, stream>>>(head_w + (size_t)l * 256 * 256 * 9,
                                            wHead + (size_t)l * 9 * 256 * 256, 256, 256, 9);
    convert_w<<<2048, 256, 0, stream>>>(head_out_w, wHOut, 254, 256, 9);
    for (int i = 0; i < 3; ++i)
        convert_w<<<2048, 256, 0, stream>>>(ref_w + (size_t)i * 128 * 256 * 9,
                                            wRef + (size_t)i * 9 * 128 * 256, 128, 256, 9);
    for (int l = 0; l < 4; ++l)
        convert_w<<<1024, 256, 0, stream>>>(tow_w + (size_t)l * 128 * 128 * 9,
                                            wTow + (size_t)l * 9 * 128 * 128, 128, 128, 9);
    convert_w<<<8, 256, 0, stream>>>(out_w, wOut, 8, 128, 1);

    size_t featOff[5]; size_t fa = 0;
    for (int i = 0; i < 5; ++i) {
        featOff[i] = fa;
        size_t tot = (size_t)2 * 256 * hw[i];
        convert_bf16<<<gblocks(tot), 256, 0, stream>>>(feat[i], featBF + fa, tot);
        fa += tot;
    }

    auto conv3 = [&](const unsigned short* in, const unsigned short* w, const float* bias,
                     float* out, int Cin, int Cout, int H, int W) {
        int nTx = (W + 7) / 8, nTy = (H + 7) / 8;
        dim3 g(nTx * nTy, (Cout + 127) / 128, 2);
        conv_wmma<3><<<g, 256, 0, stream>>>(in, w, bias, out, Cin, Cout, H, W, nTx);
    };
    auto conv1 = [&](const unsigned short* in, const unsigned short* w, const float* bias,
                     float* out, int Cin, int Cout, int H, int W) {
        int nTx = (W + 7) / 8, nTy = (H + 7) / 8;
        dim3 g(nTx * nTy, (Cout + 127) / 128, 2);
        conv_wmma<1><<<g, 256, 0, stream>>>(in, w, bias, out, Cin, Cout, H, W, nTx);
    };
    auto gn_stats = [&](const float* in, int C, int H, int W, int groups) {
        int HW = H * W, cpg = C / groups, G = 2 * groups;
        gn_partial<<<G * PBLK, 256, 0, stream>>>(in, part, C, HW, groups, cpg);
        gn_finalize<<<G, 32, 0, stream>>>(part, mr, (float)((size_t)cpg * HW));
    };
    auto gn_f32 = [&](const float* in, float* out, const float* g, const float* b,
                      int C, int H, int W, int groups) {
        gn_stats(in, C, H, W, groups);
        int HW = H * W, cpg = C / groups;
        size_t total = (size_t)2 * C * HW;
        gn_apply_f32<<<gblocks(total), 256, 0, stream>>>(in, out, g, b, mr,
                                                         C, HW, groups, cpg, total);
    };
    auto gn_bf16 = [&](const float* in, unsigned short* out, const float* g, const float* b,
                       const float* psc, int C, int H, int W, int groups) {
        gn_stats(in, C, H, W, groups);
        int HW = H * W, cpg = C / groups;
        size_t total = (size_t)2 * C * HW;
        gn_apply_bf16<<<gblocks(total), 256, 0, stream>>>(in, out, g, b, mr, psc,
                                                          C, HW, groups, cpg, total);
    };

    float* outf = (float*)d_out;
    size_t po[5]; size_t accum = 0;
    for (int i = 0; i < 5; ++i) { po[i] = accum; accum += (size_t)2 * 254 * hw[i]; }
    const size_t oMaskFeat = accum;
    const size_t oMaskLog  = oMaskFeat + (size_t)2 * 8 * 15360;

    // ---------------- shared heads on 5 FPN levels ----------------
    for (int i = 0; i < 5; ++i) {
        int H = Hh[i], W = Wz[i];
        const unsigned short* xbf = featBF + featOff[i];
        for (int l = 0; l < 4; ++l) {
            conv3(xbf, wHead + (size_t)l * 9 * 256 * 256, nullptr, convF, 256, 256, H, W);
            // layer 3: fold per-level scale into the bf16 GN output
            gn_bf16(convF, actBF, head_gn_g + l * 256, head_gn_b + l * 256,
                    (l == 3) ? (scales + i) : nullptr, 256, H, W, 32);
            xbf = actBF;
        }
        conv3(xbf, wHOut, head_out_b, outf + po[i], 256, 254, H, W);
    }

    // ---------------- mask branch ----------------
    for (int i = 0; i < 3; ++i) {
        int H = Hh[i], W = Wz[i];
        conv3(featBF + featOff[i], wRef + (size_t)i * 9 * 128 * 256, nullptr,
              maskT, 256, 128, H, W);
        if (i == 0) {
            gn_f32(maskT, mX, ref_gn_g, ref_gn_b, 128, H, W, 1);
        } else {
            gn_f32(maskT, maskT, ref_gn_g + i * 128, ref_gn_b + i * 128, 128, H, W, 1);
            resize_bilinear<<<4096, 256, 0, stream>>>(maskT, mX, 2 * 128, H, W, 96, 160, 1);
        }
    }
    {
        size_t tot = (size_t)2 * 128 * 15360;
        convert_bf16<<<gblocks(tot), 256, 0, stream>>>(mX, bfA, tot);
    }
    unsigned short* tin = bfA;
    unsigned short* tout = bfB;
    for (int l = 0; l < 4; ++l) {
        conv3(tin, wTow + (size_t)l * 9 * 128 * 128, nullptr, maskT, 128, 128, 96, 160);
        gn_bf16(maskT, tout, tow_gn_g + l * 128, tow_gn_b + l * 128, nullptr, 128, 96, 160, 1);
        unsigned short* t = tin; tin = tout; tout = t;  // activated result now in tin
    }
    conv1(tin, wOut, nullptr, mFe, 128, 8, 96, 160);
    gn_f32(mFe, outf + oMaskFeat, out_gn_g, out_gn_b, 8, 96, 160, 1);

    // ---------------- dynamic conv + 2x upsample ----------------
    dyn_conv<<<2 * N_INST, 256, 0, stream>>>(outf + oMaskFeat, outf + po[0],
                                             mLog, 96, 160);
    resize_bilinear<<<8192, 256, 0, stream>>>(mLog, outf + oMaskLog,
                                              200, 96, 160, 192, 320, 0);
}
